// RawGlobalExtractor_13254269075921
// MI455X (gfx1250) — matrix operand order repaired and checked by reading, not compiled, not run on hardware
//
#include <hip/hip_runtime.h>

#define HIDDEN 1024
#define SEQ    512
#define BATCH  16
#define NENT   9
#define NARG   12
#define NEGV   1.0e12f
#define LOG2_10000 13.287712379549449f

typedef __attribute__((ext_vector_type(16))) __bf16 v16bf;
typedef __attribute__((ext_vector_type(8)))  float  v8f;

union FragBF {
    v16bf v;
    uint4 u[2];
};

__device__ __forceinline__ v8f wmma_bf16(v16bf a, v16bf b, v8f c) {
    // D = A x B + C, 16x16x32 bf16 -> f32 accum
    return __builtin_amdgcn_wmma_f32_16x16x32_bf16(
        /*neg_a=*/false, a, /*neg_b=*/false, b,
        /*c_mod=*/(short)0, c, /*reuse_a=*/false, /*reuse_b=*/false);
}

// ---------------------------------------------------------------------------
// f32 -> bf16 streaming convert
// ---------------------------------------------------------------------------
__global__ void cvt_f32_bf16(const float* __restrict__ in,
                             __bf16* __restrict__ out, int n) {
    int i = blockIdx.x * blockDim.x + threadIdx.x;
    int stride = gridDim.x * blockDim.x;
    for (; i < n; i += stride) out[i] = (__bf16)in[i];
}

// ---------------------------------------------------------------------------
// Stage 1: P = H @ W + b, then RoPE; write Q row-major [bh][s][64] and
// K transposed [bh][64][s] (both bf16). One wave computes a 32(M) x 64(N)
// tile (8 accumulators); B fragments are reused across the two M-subtiles.
// ---------------------------------------------------------------------------
__global__ void __launch_bounds__(128)
gemm_rope_kernel(const __bf16* __restrict__ Hbf,
                 const __bf16* __restrict__ Wbf,
                 const float*  __restrict__ bias,
                 __bf16* __restrict__ Qout,
                 __bf16* __restrict__ Ktout,
                 int Nout, int NT, int ntilesN)
{
    const int lane = threadIdx.x & 31;
    const int wid  = (blockIdx.x * blockDim.x + threadIdx.x) >> 5;
    const int mt   = wid / ntilesN;
    const int nt   = wid % ntilesN;
    const int row0 = mt * 32;            // flattened b*SEQ + s
    const int n0   = nt * 64;

    const int lm = lane & 15;            // M (A) / N (B,C,D) index
    const int lh = lane >> 4;            // lane half
    const int kk = lm + 16 * lh;         // B-fragment K row within 32-chunk

    const __bf16* arow0 = Hbf + (size_t)(row0 + lm) * HIDDEN + lh * 8;
    const __bf16* arow1 = arow0 + (size_t)16 * HIDDEN;

    v8f c[2][4] = {};

    for (int k0 = 0; k0 < HIDDEN; k0 += 32) {
        FragBF a0, a1;
        a0.u[0] = *(const uint4*)(arow0 + k0);
        a0.u[1] = *(const uint4*)(arow0 + k0 + 16);
        a1.u[0] = *(const uint4*)(arow1 + k0);
        a1.u[1] = *(const uint4*)(arow1 + k0 + 16);
        const __bf16* brow = Wbf + (size_t)(k0 + lh * 8) * Nout + n0 + lm;
#pragma unroll
        for (int i = 0; i < 4; ++i) {
            FragBF bf;
#pragma unroll
            for (int s = 0; s < 16; ++s)
                bf.v[s] = brow[(size_t)(s < 8 ? s : s + 8) * Nout + 16 * i];
            c[0][i] = wmma_bf16(a0.v, bf.v, c[0][i]);
            c[1][i] = wmma_bf16(a1.v, bf.v, c[1][i]);
        }
    }

    // Epilogue: bias + RoPE, scatter to Q / Kt (bf16)
    const int bidx = row0 / SEQ;
#pragma unroll
    for (int mi = 0; mi < 2; ++mi) {
        const int sbase = (row0 % SEQ) + mi * 16;
#pragma unroll
        for (int i = 0; i < 4; ++i) {
            const int n = n0 + 16 * i + lm;          // output column
            const float bs = bias[n];
            const int d   = n & 127;                 // index within [q|k] block
            const int h   = n >> 7;                  // head
            const int j   = d & 63;                  // dim within 64-d head
            const int isq = (d < 64);
            // inv_freq = 10000^(-2*(j/2)/64) = 2^(-(j/2)*(2/64)*log2(10000))
            const float inv = exp2f(-(float)(j >> 1) * (2.0f / 64.0f) * LOG2_10000);
            const float sgn = (j & 1) ? 1.0f : -1.0f;
            const size_t bh = (size_t)bidx * NT + h;
#pragma unroll
            for (int r = 0; r < 8; ++r) {
                const int srow = sbase + r + lh * 8; // sequence position
                float val = c[mi][i][r] + bs;
                float partner = __shfl_xor(val, 1, 32); // RoPE pair in lane^1
                const float ang = (float)srow * inv;
                const float sn = __sinf(ang);
                const float cs = __cosf(ang);
                const float rot = val * cs + sgn * partner * sn;
                if (isq) Qout [(bh * SEQ + srow) * 64 + j] = (__bf16)rot;
                else     Ktout[(bh * 64 + j) * SEQ + srow] = (__bf16)rot;
            }
        }
    }
}

// ---------------------------------------------------------------------------
// Stage 2: logits[bh][m][n] = (Q[bh] . K[bh]^T) masked / 8.
// One wave computes a 32(M) x 64(N) tile; A fragments loaded once and held,
// B fragments reused across the two M-subtiles. K dim = 64 -> 2 chunks.
// ---------------------------------------------------------------------------
__global__ void __launch_bounds__(256)
attn_logits_kernel(const __bf16* __restrict__ Q,    // [BH][SEQ][64]
                   const __bf16* __restrict__ Kt,   // [BH][64][SEQ]
                   const float*  __restrict__ mask, // [B][SEQ]
                   float* __restrict__ out,         // [BH][SEQ][SEQ]
                   int NT)
{
    const int lane = threadIdx.x & 31;
    const int wid  = (blockIdx.x * blockDim.x + threadIdx.x) >> 5;
    const int NTILE = SEQ / 64;   // 8
    const int MTILE = SEQ / 32;   // 16
    const int bh  = wid / (MTILE * NTILE);
    const int rem = wid % (MTILE * NTILE);
    const int mt  = rem / NTILE;
    const int nt  = rem % NTILE;
    const int m0  = mt * 32, n0 = nt * 64;
    const int lm  = lane & 15, lh = lane >> 4;
    const int kk  = lm + 16 * lh;

    // A fragments: Q rows m0+lm (+16), 64-d split into two 32-chunks
    const __bf16* qrow = Q + ((size_t)bh * SEQ + m0 + lm) * 64 + lh * 8;
    FragBF a[2][2];
#pragma unroll
    for (int mi = 0; mi < 2; ++mi) {
#pragma unroll
        for (int kc = 0; kc < 2; ++kc) {
            const __bf16* p = qrow + mi * 16 * 64 + kc * 32;
            a[mi][kc].u[0] = *(const uint4*)(p);
            a[mi][kc].u[1] = *(const uint4*)(p + 16);
        }
    }

    v8f c[2][4] = {};
#pragma unroll
    for (int kc = 0; kc < 2; ++kc) {
        const __bf16* krow = Kt + ((size_t)bh * 64 + kc * 32 + lh * 8) * SEQ + n0 + lm;
#pragma unroll
        for (int i = 0; i < 4; ++i) {
            FragBF bf;
#pragma unroll
            for (int s = 0; s < 16; ++s)
                bf.v[s] = krow[(size_t)(s < 8 ? s : s + 8) * SEQ + 16 * i];
            c[0][i] = wmma_bf16(a[0][kc].v, bf.v, c[0][i]);
            c[1][i] = wmma_bf16(a[1][kc].v, bf.v, c[1][i]);
        }
    }

    const int bidx = bh / NT;
#pragma unroll
    for (int i = 0; i < 4; ++i) {
        const int n = n0 + 16 * i + lm;
        const float pad    = mask[bidx * SEQ + n];
        const float padneg = (1.0f - pad) * NEGV;
#pragma unroll
        for (int mi = 0; mi < 2; ++mi) {
#pragma unroll
            for (int r = 0; r < 8; ++r) {
                const int m = m0 + mi * 16 + r + lh * 8;
                float v = c[mi][i][r] * pad - padneg;
                if (m > n) v -= NEGV;                 // strict lower-tri mask
                out[((size_t)bh * SEQ + m) * SEQ + n] = v * 0.125f;
            }
        }
    }
}

// ---------------------------------------------------------------------------
extern "C" void kernel_launch(void* const* d_in, const int* in_sizes, int n_in,
                              void* d_out, int out_size, void* d_ws, size_t ws_size,
                              hipStream_t stream) {
    (void)in_sizes; (void)n_in; (void)out_size; (void)ws_size;

    const float* H   = (const float*)d_in[0]; // [16,512,1024]
    const float* dw  = (const float*)d_in[1]; // [1024, 1152]
    const float* db  = (const float*)d_in[2]; // [1152]
    const float* aw  = (const float*)d_in[3]; // [1024, 1536]
    const float* ab  = (const float*)d_in[4]; // [1536]
    const float* msk = (const float*)d_in[5]; // [16, 512]
    float* out = (float*)d_out;

    char* ws = (char*)d_ws;
    __bf16* Hbf   = (__bf16*)ws; ws += (size_t)BATCH * SEQ * HIDDEN * 2;
    __bf16* Wentb = (__bf16*)ws; ws += (size_t)HIDDEN * (NENT * 128) * 2;
    __bf16* Wargb = (__bf16*)ws; ws += (size_t)HIDDEN * (NARG * 128) * 2;
    __bf16* Qent  = (__bf16*)ws; ws += (size_t)BATCH * NENT * SEQ * 64 * 2;
    __bf16* Ktent = (__bf16*)ws; ws += (size_t)BATCH * NENT * SEQ * 64 * 2;
    __bf16* Qarg  = (__bf16*)ws; ws += (size_t)BATCH * NARG * SEQ * 64 * 2;
    __bf16* Ktarg = (__bf16*)ws; ws += (size_t)BATCH * NARG * SEQ * 64 * 2;

    cvt_f32_bf16<<<2048, 256, 0, stream>>>(H,  Hbf,   BATCH * SEQ * HIDDEN);
    cvt_f32_bf16<<<512,  256, 0, stream>>>(dw, Wentb, HIDDEN * NENT * 128);
    cvt_f32_bf16<<<512,  256, 0, stream>>>(aw, Wargb, HIDDEN * NARG * 128);

    {   // ent head projection + RoPE
        const int ntilesN = (NENT * 128) / 64;                  // 18
        const int waves = (BATCH * SEQ / 32) * ntilesN;         // 4608
        gemm_rope_kernel<<<waves / 4, 128, 0, stream>>>(
            Hbf, Wentb, db, Qent, Ktent, NENT * 128, NENT, ntilesN);
    }
    {   // arg head projection + RoPE
        const int ntilesN = (NARG * 128) / 64;                  // 24
        const int waves = (BATCH * SEQ / 32) * ntilesN;         // 6144
        gemm_rope_kernel<<<waves / 4, 128, 0, stream>>>(
            Hbf, Wargb, ab, Qarg, Ktarg, NARG * 128, NARG, ntilesN);
    }
    {   // ent logits
        const int waves = BATCH * NENT * (SEQ / 32) * (SEQ / 64); // 18432
        attn_logits_kernel<<<waves / 8, 256, 0, stream>>>(
            Qent, Ktent, msk, out, NENT);
    }
    {   // arg logits
        const int waves = BATCH * NARG * (SEQ / 32) * (SEQ / 64); // 24576
        attn_logits_kernel<<<waves / 8, 256, 0, stream>>>(
            Qarg, Ktarg, msk, out + (size_t)BATCH * NENT * SEQ * SEQ, NARG);
    }
}
